// Spatialatt_26594437496901
// MI455X (gfx1250) — matrix operand order repaired and checked by reading, not compiled, not run on hardware
//
#include <hip/hip_runtime.h>
#include <hip/hip_bf16.h>

// ---------------------------------------------------------------------------
// MI455X (gfx1250) implementation.
//  - All large GEMMs via v_wmma_f32_16x16x32_bf16 (fp32 accumulate).
//  - Global->LDS tiles moved with GLOBAL_LOAD_ASYNC_TO_LDS_B128 (ASYNCcnt,
//    double-buffered, zero VGPR staging).
//  - QKV computed once and shared across all cliques (algebraic dedup).
//  - Tiny attention / LN / epilogues in fp32 VALU.
// ---------------------------------------------------------------------------

typedef __bf16  bf16x16 __attribute__((ext_vector_type(16)));
typedef __bf16  bf16x8  __attribute__((ext_vector_type(8)));
typedef float   v8f     __attribute__((ext_vector_type(8)));

#define M_ROWS 15360   // B*T*N = 64*20*12
#define DDIM   1024
#define ATT_SCALE 0.03125f   // 1/sqrt(1024)
#define LN_EPS 1e-5f

__device__ __forceinline__ unsigned short f2bf(float x) {
    union { float f; unsigned u; } v; v.f = x;
    unsigned r = v.u + 0x7FFFu + ((v.u >> 16) & 1u);   // round-to-nearest-even
    return (unsigned short)(r >> 16);
}
__device__ __forceinline__ float bf2f(unsigned short u) {
    union { unsigned u; float f; } v; v.u = ((unsigned)u) << 16;
    return v.f;
}

// Async global -> LDS copy of 16 bytes (GLOBAL_LOAD_ASYNC_TO_LDS_B128).
// lds  : 32-bit LDS byte address (per-lane)
// gaddr: 64-bit global byte address (per-lane)
__device__ __forceinline__ void async_ld_b128(unsigned lds, unsigned long long gaddr) {
    asm volatile("global_load_async_to_lds_b128 %0, %1, off"
                 :: "v"(lds), "v"(gaddr) : "memory");
}
__device__ __forceinline__ void wait_async_le8() {
    asm volatile("s_wait_asynccnt 0x8" ::: "memory");
}
__device__ __forceinline__ void wait_async_0() {
    asm volatile("s_wait_asynccnt 0x0" ::: "memory");
}

// ---------------------------------------------------------------------------
// fp32 -> bf16 cast (grid-stride)
// ---------------------------------------------------------------------------
__global__ void cast_f32_bf16(const float* __restrict__ in,
                              unsigned short* __restrict__ out, int n) {
    for (int i = blockIdx.x * blockDim.x + threadIdx.x; i < n;
         i += gridDim.x * blockDim.x)
        out[i] = f2bf(in[i]);
}

// ---------------------------------------------------------------------------
// Tiled bf16 WMMA GEMM:  C[M,N] = A[M,K] * B[N,K]^T  (both K-major)
// Block 128 threads (4 wave32), block tile 128x128, wave tile 64x64,
// K-step 32 (one WMMA depth), double-buffered LDS filled by async copies.
// Epilogue: optional bias (per-col), ReLU, scale, accumulate; bf16/f32 out.
// ---------------------------------------------------------------------------
union FragAB { bf16x16 v; bf16x8 h[2]; };

#define LDS_STRIDE 40            // halves per row (padded, conflict-free b128)
#define LDS_ROWB   (LDS_STRIDE * 2)      // 80 bytes
#define LDS_BUFB   (128 * LDS_ROWB)      // 10240 bytes per buffer

template <int OUT_BF16, int RELU, int ACCUM>
__global__ __launch_bounds__(128)
void gemm_bf16_wmma(const unsigned short* __restrict__ A,
                    const unsigned short* __restrict__ Bw,
                    void* __restrict__ Cout,
                    const float* __restrict__ bias,
                    const float* __restrict__ scale_ptr,
                    int Mdim, int Ndim, int Kdim) {
    __shared__ alignas(16) __bf16 As[2][128][LDS_STRIDE];
    __shared__ alignas(16) __bf16 Bs[2][128][LDS_STRIDE];

    const int bm0 = blockIdx.x * 128;
    const int bn0 = blockIdx.y * 128;

    const int lane = threadIdx.x & 31;
    const int wave = threadIdx.x >> 5;          // 0..3
    const int m0   = (wave & 1) * 64;           // wave tile 64x64
    const int n0   = (wave >> 1) * 64;

    const int mrow  = lane & 15;
    const int khalf = lane >> 4;

    const float scl = scale_ptr ? scale_ptr[0] : 1.0f;

    // cooperative loaders: thread t owns row t of both tiles (32 halves = 4xb128)
    const int t = threadIdx.x;                  // 0..127
    const unsigned long long gA =
        (unsigned long long)(A  + (size_t)(bm0 + t) * Kdim);
    const unsigned long long gB =
        (unsigned long long)(Bw + (size_t)(bn0 + t) * Kdim);
    const unsigned ldsA = (unsigned)(size_t)&As[0][t][0];
    const unsigned ldsB = (unsigned)(size_t)&Bs[0][t][0];

    v8f acc[4][4];
    const v8f z8 = {0.f,0.f,0.f,0.f,0.f,0.f,0.f,0.f};
    for (int i = 0; i < 4; ++i)
        for (int j = 0; j < 4; ++j) acc[i][j] = z8;

    const int NKT = Kdim >> 5;                  // K steps of 32

    // prologue: fill buffer 0 (8 async b128 per thread / per wave)
    {
        const unsigned long long goff = 0ull;
        #pragma unroll
        for (int j = 0; j < 4; ++j) {
            async_ld_b128(ldsA + j * 16, gA + goff + (unsigned)(j * 16));
            async_ld_b128(ldsB + j * 16, gB + goff + (unsigned)(j * 16));
        }
    }

    for (int kt = 0; kt < NKT; ++kt) {
        const int buf = kt & 1;
        if (kt + 1 < NKT) {
            // issue next tile into the other buffer (safe: last read at kt-1,
            // all waves passed the trailing barrier of iteration kt-1)
            const unsigned nb = (unsigned)((kt + 1) & 1) * LDS_BUFB;
            const unsigned long long goff = (unsigned long long)((kt + 1) << 6); // *32 halves *2B
            #pragma unroll
            for (int j = 0; j < 4; ++j) {
                async_ld_b128(ldsA + nb + j * 16, gA + goff + (unsigned)(j * 16));
                async_ld_b128(ldsB + nb + j * 16, gB + goff + (unsigned)(j * 16));
            }
            if (kt + 2 < NKT) {                 // -> global_prefetch_b8
                __builtin_prefetch((const void*)(gA + (unsigned long long)((kt + 2) << 6)), 0, 0);
                __builtin_prefetch((const void*)(gB + (unsigned long long)((kt + 2) << 6)), 0, 0);
            }
            wait_async_le8();                   // current buffer complete (in-order)
        } else {
            wait_async_0();
        }
        __syncthreads();                        // all waves' portions visible

        // Fragments (batched loads -> single dscnt wait -> 16 back-to-back WMMA)
        // A layout: lane(m,half): e0..7 -> K=8*half+e, e8..15 -> 16+8*half+(e-8)
        // B layout: lane(n,khalf): e0..7 -> K=8*khalf+e, e8..15 -> 16+8*khalf+(e-8)
        FragAB fa[4], fb[4];
        #pragma unroll
        for (int mt = 0; mt < 4; ++mt) {
            const int r = m0 + mt * 16 + mrow;
            fa[mt].h[0] = *(const bf16x8*)&As[buf][r][khalf * 8];
            fa[mt].h[1] = *(const bf16x8*)&As[buf][r][16 + khalf * 8];
        }
        #pragma unroll
        for (int nt = 0; nt < 4; ++nt) {
            const int r = n0 + nt * 16 + mrow;
            fb[nt].h[0] = *(const bf16x8*)&Bs[buf][r][khalf * 8];
            fb[nt].h[1] = *(const bf16x8*)&Bs[buf][r][16 + khalf * 8];
        }
        #pragma unroll
        for (int mt = 0; mt < 4; ++mt)
            #pragma unroll
            for (int nt = 0; nt < 4; ++nt)
                acc[mt][nt] = __builtin_amdgcn_wmma_f32_16x16x32_bf16(
                    false, fa[mt].v, false, fb[nt].v,
                    (short)0, acc[mt][nt], false, false);

        __syncthreads();                        // done reading buf before overwrite
    }

    // Epilogue. C layout: VGPR r -> row (r + 8*(lane>>4)), col = lane&15.
    const int cN = lane & 15;
    const int cM = 8 * (lane >> 4);
    for (int mt = 0; mt < 4; ++mt) {
        for (int nt = 0; nt < 4; ++nt) {
            const int col = bn0 + n0 + nt * 16 + cN;
            const float bv = bias ? bias[col] : 0.0f;
            for (int r = 0; r < 8; ++r) {
                const int row = bm0 + m0 + mt * 16 + cM + r;
                float v = acc[mt][nt][r] + bv;
                if (RELU) v = v > 0.0f ? v : 0.0f;
                v *= scl;
                const size_t idx = (size_t)row * Ndim + col;
                if (OUT_BF16) {
                    ((unsigned short*)Cout)[idx] = f2bf(v);
                } else {
                    float* o = (float*)Cout;
                    if (ACCUM) o[idx] += v; else o[idx] = v;
                }
            }
        }
    }
}

// ---------------------------------------------------------------------------
// Softmax-free clique attention. One block (128 threads) per (group, head).
// AX[row, h*128+l] = sum_e (scale * <Q[ci],K[e]>) * V[e][l]
// ---------------------------------------------------------------------------
__global__ __launch_bounds__(128)
void attention_kernel(const unsigned short* __restrict__ Q,
                      const unsigned short* __restrict__ K,
                      const unsigned short* __restrict__ V,
                      unsigned short* __restrict__ AX, int c) {
    __shared__ float Qs[12][128];
    __shared__ float Ks[12][128];
    __shared__ float Vs[12][128];
    __shared__ float att[12][12];

    const int g = blockIdx.x >> 3;
    const int h = blockIdx.x & 7;
    const int l = threadIdx.x;
    const int colbase = h * 128;

    for (int e = 0; e < c; ++e) {
        const size_t off = (size_t)(g * c + e) * DDIM + colbase + l;
        Qs[e][l] = bf2f(Q[off]);
        Ks[e][l] = bf2f(K[off]);
        Vs[e][l] = bf2f(V[off]);
    }
    __syncthreads();

    for (int p = threadIdx.x; p < c * c; p += 128) {
        const int ci = p / c, e = p % c;
        float s = 0.0f;
        for (int j = 0; j < 128; ++j) s += Qs[ci][j] * Ks[e][j];
        att[ci][e] = s * ATT_SCALE;
    }
    __syncthreads();

    for (int ci = 0; ci < c; ++ci) {
        float s = 0.0f;
        for (int e = 0; e < c; ++e) s += att[ci][e] * Vs[e][l];
        AX[(size_t)(g * c + ci) * DDIM + colbase + l] = f2bf(s);
    }
}

// ---------------------------------------------------------------------------
// Residual + LayerNorm, emits bf16 FFN input. One block (256) per row.
// ---------------------------------------------------------------------------
__global__ __launch_bounds__(256)
void ln_kernel(const float* __restrict__ Y0, const float* __restrict__ feat,
               const float* __restrict__ g, const float* __restrict__ b,
               unsigned short* __restrict__ out) {
    __shared__ float sh0[256];
    __shared__ float sh1[256];
    const int row = blockIdx.x;
    const int tid = threadIdx.x;
    const size_t base = (size_t)row * DDIM;

    float v[4];
    float s = 0.0f, s2 = 0.0f;
    for (int j = 0; j < 4; ++j) {
        const int col = tid + j * 256;
        const float x = Y0[base + col] + feat[base + col];
        v[j] = x; s += x; s2 += x * x;
    }
    sh0[tid] = s; sh1[tid] = s2;
    __syncthreads();
    for (int off = 128; off > 0; off >>= 1) {
        if (tid < off) { sh0[tid] += sh0[tid + off]; sh1[tid] += sh1[tid + off]; }
        __syncthreads();
    }
    const float mu  = sh0[0] * (1.0f / DDIM);
    const float var = sh1[0] * (1.0f / DDIM) - mu * mu;
    const float rr  = rsqrtf(var + LN_EPS);
    for (int j = 0; j < 4; ++j) {
        const int col = tid + j * 256;
        out[base + col] = f2bf((v[j] - mu) * rr * g[col] + b[col]);
    }
}

// ---------------------------------------------------------------------------
// Host launcher
// ---------------------------------------------------------------------------
extern "C" void kernel_launch(void* const* d_in, const int* in_sizes, int n_in,
                              void* d_out, int out_size, void* d_ws, size_t ws_size,
                              hipStream_t stream) {
    const float* features = (const float*)d_in[0];
    const float* Wq       = (const float*)d_in[1];
    const float* Wk       = (const float*)d_in[2];
    const float* Wv       = (const float*)d_in[3];
    const float* W_out    = (const float*)d_in[4];
    const float* ln_g     = (const float*)d_in[5];
    const float* ln_b     = (const float*)d_in[6];
    const float* ffn1_w   = (const float*)d_in[7];
    const float* ffn1_b   = (const float*)d_in[8];
    const float* ffn2_w   = (const float*)d_in[9];
    const float* ffn2_b   = (const float*)d_in[10];
    const float* w_clique = (const float*)d_in[11];
    float* out = (float*)d_out;

    const size_t mb = (size_t)M_ROWS * DDIM;   // 15,728,640 elements
    const size_t wb = (size_t)DDIM * DDIM;     // 1,048,576 elements

    unsigned short* Xbf  = (unsigned short*)d_ws;
    unsigned short* Qbf  = Xbf + mb;
    unsigned short* Kbf  = Qbf + mb;
    unsigned short* Vbf  = Kbf + mb;
    unsigned short* AXbf = Vbf + mb;
    unsigned short* Ybf  = AXbf + mb;
    unsigned short* Hbf  = Ybf + mb;
    float*          Yf32 = (float*)(Hbf + mb);
    unsigned short* Wqb  = (unsigned short*)(Yf32 + mb);
    unsigned short* Wkb  = Wqb + wb;
    unsigned short* Wvb  = Wkb + wb;
    unsigned short* Wob  = Wvb + wb;
    unsigned short* F1b  = Wob + wb;
    unsigned short* F2b  = F1b + wb;

    // 1) casts
    cast_f32_bf16<<<dim3((int)((mb + 1023) / 1024)), 256, 0, stream>>>(features, Xbf, (int)mb);
    cast_f32_bf16<<<dim3((int)((wb + 1023) / 1024)), 256, 0, stream>>>(Wq,     Wqb, (int)wb);
    cast_f32_bf16<<<dim3((int)((wb + 1023) / 1024)), 256, 0, stream>>>(Wk,     Wkb, (int)wb);
    cast_f32_bf16<<<dim3((int)((wb + 1023) / 1024)), 256, 0, stream>>>(Wv,     Wvb, (int)wb);
    cast_f32_bf16<<<dim3((int)((wb + 1023) / 1024)), 256, 0, stream>>>(W_out,  Wob, (int)wb);
    cast_f32_bf16<<<dim3((int)((wb + 1023) / 1024)), 256, 0, stream>>>(ffn1_w, F1b, (int)wb);
    cast_f32_bf16<<<dim3((int)((wb + 1023) / 1024)), 256, 0, stream>>>(ffn2_w, F2b, (int)wb);

    const dim3 ggrid(M_ROWS / 128, DDIM / 128);   // 120 x 8
    const dim3 gblk(128);                         // 4 wave32

    // 2) QKV projections (shared across all cliques)
    gemm_bf16_wmma<1,0,0><<<ggrid, gblk, 0, stream>>>(Xbf, Wqb, Qbf, nullptr, nullptr, M_ROWS, DDIM, DDIM);
    gemm_bf16_wmma<1,0,0><<<ggrid, gblk, 0, stream>>>(Xbf, Wkb, Kbf, nullptr, nullptr, M_ROWS, DDIM, DDIM);
    gemm_bf16_wmma<1,0,0><<<ggrid, gblk, 0, stream>>>(Xbf, Wvb, Vbf, nullptr, nullptr, M_ROWS, DDIM, DDIM);

    // 3) per-clique pipeline, accumulating weighted FFN output into d_out
    const int cliques[4] = {2, 3, 4, 12};
    for (int ci = 0; ci < 4; ++ci) {
        const int c = cliques[ci];
        const int G = M_ROWS / c;

        attention_kernel<<<dim3(G * 8), dim3(128), 0, stream>>>(Qbf, Kbf, Vbf, AXbf, c);

        gemm_bf16_wmma<0,0,0><<<ggrid, gblk, 0, stream>>>(AXbf, Wob, Yf32, nullptr, nullptr, M_ROWS, DDIM, DDIM);

        ln_kernel<<<dim3(M_ROWS), dim3(256), 0, stream>>>(Yf32, features, ln_g, ln_b, Ybf);

        gemm_bf16_wmma<1,1,0><<<ggrid, gblk, 0, stream>>>(Ybf, F1b, Hbf, ffn1_b, nullptr, M_ROWS, DDIM, DDIM);

        if (ci == 0)
            gemm_bf16_wmma<0,0,0><<<ggrid, gblk, 0, stream>>>(Hbf, F2b, out, ffn2_b, w_clique + ci, M_ROWS, DDIM, DDIM);
        else
            gemm_bf16_wmma<0,0,1><<<ggrid, gblk, 0, stream>>>(Hbf, F2b, out, ffn2_b, w_clique + ci, M_ROWS, DDIM, DDIM);
    }
}
